// GroupQueryAttention_71734543778306
// MI455X (gfx1250) — compile-verified, matrix operand order checked by reading, not run on hardware
//
#include <hip/hip_runtime.h>
#include <hip/hip_bf16.h>

typedef __bf16 bf16;
typedef __attribute__((ext_vector_type(16))) __bf16 v16bf;
typedef __attribute__((ext_vector_type(8)))  __bf16 v8bf;
typedef __attribute__((ext_vector_type(4)))  __bf16 v4bf;
typedef __attribute__((ext_vector_type(8)))  float  v8f;

#define WMMA_BF16(a, b, c) \
  __builtin_amdgcn_wmma_f32_16x16x32_bf16(false, (a), false, (b), (short)0, (c), false, false)

// Problem constants (match reference)
#define Bc   2
#define Sc   2048
#define Hc   2048
#define NHc  16
#define Gc   4
#define HDc  128
#define KVW  (Gc * HDc)   // 512

// Load one 16-bit WMMA operand fragment from a K-major row: lane reads
// elements [0..7] and [16..23] relative to p (p already includes the
// per-lane kbase = (lane>>4)*8 offset). Both chunks are 16B aligned.
__device__ __forceinline__ v16bf ldfrag(const bf16* p) {
  v8bf lo = *(const v8bf*)(p);
  v8bf hi = *(const v8bf*)(p + 16);
  return __builtin_shufflevector(lo, hi, 0, 1, 2, 3, 4, 5, 6, 7,
                                         8, 9, 10, 11, 12, 13, 14, 15);
}

// Stage 4 contiguous elements into LDS as bf16 (f32 input -> convert; bf16 -> copy)
__device__ __forceinline__ void loadcvt4(const float* g, bf16* s) {
  float4 t = *(const float4*)g;
  v4bf o;
  o[0] = (bf16)t.x; o[1] = (bf16)t.y; o[2] = (bf16)t.z; o[3] = (bf16)t.w;
  *(v4bf*)s = o;
}
__device__ __forceinline__ void loadcvt4(const bf16* g, bf16* s) {
  *(v4bf*)s = *(const v4bf*)g;
}

__device__ __forceinline__ void store_out(float* p, float v) { *p = v; }
__device__ __forceinline__ void store_out(bf16* p, float v) { *p = (bf16)v; }

// ---------------------------------------------------------------------------
// Tiled GEMM:  C[M][N] = A[M][K] @ W[K][N] + bias.
// BM=128 BN=128 BK=32, 256 threads = 8 waves (4 M-waves x 2 N-waves),
// each wave computes 32x64 = 8 WMMA tiles per K-step (8 wmma : 12 ds_load_b128)
// ---------------------------------------------------------------------------
template <typename AT, typename OT>
__global__ __launch_bounds__(256) void gemm_wmma(const AT* __restrict__ A,
                                                 const float* __restrict__ W,
                                                 const float* __restrict__ bias,
                                                 OT* __restrict__ C,
                                                 int N, int K) {
  __shared__ __align__(16) bf16 Als[128][32];   // [m][k]       8 KB
  __shared__ __align__(16) bf16 Bls[128][32];   // W^T: [n][k]  8 KB

  const int tid   = threadIdx.x;
  const int wave  = tid >> 5;
  const int lane  = tid & 31;
  const int lrow  = lane & 15;
  const int kbase = (lane >> 4) << 3;
  const int half8 = kbase;                  // 0 or 8: accumulator row offset
  const int mwave = wave & 3;               // 4 M-waves: 32 rows each
  const int nwave = wave >> 2;              // 2 N-waves: 64 cols each
  const int m0    = blockIdx.y * 128;
  const int n0    = blockIdx.x * 128;

  v8f acc[2][4];
#pragma unroll
  for (int t = 0; t < 2; ++t)
#pragma unroll
    for (int j = 0; j < 4; ++j)
#pragma unroll
      for (int v = 0; v < 8; ++v) acc[t][j][v] = 0.0f;

  for (int k0 = 0; k0 < K; k0 += 32) {
    // Stage A tile 128x32 (4 elems per thread-iteration, 4 iterations)
#pragma unroll
    for (int i = tid; i < 1024; i += 256) {
      int r = i >> 3;
      int c = (i & 7) << 2;
      loadcvt4(&A[(size_t)(m0 + r) * K + k0 + c], &Als[r][c]);
    }
    // Stage W tile 32x128 transposed into Bls[n][k]
#pragma unroll
    for (int i = tid; i < 1024; i += 256) {
      int kr = i >> 5;
      int c4 = (i & 31) << 2;
      float4 w = *(const float4*)&W[(size_t)(k0 + kr) * N + n0 + c4];
      Bls[c4 + 0][kr] = (bf16)w.x;
      Bls[c4 + 1][kr] = (bf16)w.y;
      Bls[c4 + 2][kr] = (bf16)w.z;
      Bls[c4 + 3][kr] = (bf16)w.w;
    }
    if (k0 + 32 < K)  // hint next A tile toward the caches
      __builtin_prefetch(&A[(size_t)(m0 + (tid >> 1)) * K + k0 + 32], 0, 0);
    __syncthreads();

    v16bf af0 = ldfrag(&Als[mwave * 32 + lrow][kbase]);
    v16bf af1 = ldfrag(&Als[mwave * 32 + 16 + lrow][kbase]);
#pragma unroll
    for (int j = 0; j < 4; ++j) {
      v16bf bfrag = ldfrag(&Bls[nwave * 64 + j * 16 + lrow][kbase]);
      acc[0][j] = WMMA_BF16(af0, bfrag, acc[0][j]);
      acc[1][j] = WMMA_BF16(af1, bfrag, acc[1][j]);
    }
    __syncthreads();
  }

#pragma unroll
  for (int t = 0; t < 2; ++t) {
#pragma unroll
    for (int j = 0; j < 4; ++j) {
      int col  = n0 + nwave * 64 + j * 16 + lrow;
      float bz = bias[col];
#pragma unroll
      for (int v = 0; v < 8; ++v) {
        int row = m0 + mwave * 32 + t * 16 + v + half8;
        store_out(&C[(size_t)row * N + col], acc[t][j][v] + bz);
      }
    }
  }
}

// ---------------------------------------------------------------------------
// Flash attention (causal, GQA). One WG = 64 query rows of one (b, head).
// 4 waves; each wave: 16 query rows. Key blocks of 32.
// K tile staged via async global->LDS DMA (ASYNCcnt); V staged with transpose.
// ---------------------------------------------------------------------------
__global__ __launch_bounds__(128) void attn_wmma(const bf16* __restrict__ Qb,
                                                 const bf16* __restrict__ Kb,
                                                 const bf16* __restrict__ Vb,
                                                 bf16* __restrict__ AO) {
  __shared__ __align__(16) bf16 Kls[32][HDc];     // [t][d]   8 KB
  __shared__ __align__(16) bf16 Vtls[HDc][32];    // [d][t]   8 KB
  __shared__ __align__(16) bf16 Pls[4][16][32];   // per-wave P tile  4 KB

  const int tid   = threadIdx.x;
  const int wave  = tid >> 5;
  const int lane  = tid & 31;
  const int lrow  = lane & 15;
  const int kbase = (lane >> 4) << 3;
  const int half8 = kbase;

  const int bh = blockIdx.y;
  const int b  = bh / NHc;
  const int h  = bh % NHc;
  const int g  = h >> 2;                 // h / (NH/G)
  const int q0 = blockIdx.x * 64;
  const int qw = q0 + wave * 16;

  const float scale = 0.08838834764831845f;  // 1/sqrt(128)

  // Q fragments for this wave's 16 rows, all 4 K-chunks (HD=128 = 4x32)
  v16bf qf[4];
  {
    const bf16* qrow = Qb + ((size_t)(b * Sc + qw + lrow) * Hc + h * HDc);
#pragma unroll
    for (int c = 0; c < 4; ++c) qf[c] = ldfrag(qrow + c * 32 + kbase);
  }

  v8f oacc[8];
#pragma unroll
  for (int jt = 0; jt < 8; ++jt)
#pragma unroll
    for (int v = 0; v < 8; ++v) oacc[jt][v] = 0.0f;
  float mrow[8], lsum[8];
#pragma unroll
  for (int v = 0; v < 8; ++v) { mrow[v] = -1e30f; lsum[v] = 0.0f; }

  const int nkb = q0 / 32 + 2;  // covers keys up to q0+63 (causal)
  for (int kb = 0; kb < nkb; ++kb) {
    const int t0 = kb * 32;
    // Stage K block [32][128] via async global->LDS b128 copies (ASYNCcnt)
#pragma unroll
    for (int i = tid; i < 512; i += 128) {
      int r = i >> 4;
      int c = (i & 15) << 3;
      unsigned lds_off = (unsigned)(uintptr_t)&Kls[r][c];
      const bf16* gp = Kb + ((size_t)(b * Sc + t0 + r) * KVW + g * HDc + c);
      asm volatile("global_load_async_to_lds_b128 %0, %1, off"
                   :: "v"(lds_off), "v"(gp)
                   : "memory");
    }
    // Stage V block transposed -> Vtls[d][t]
#pragma unroll
    for (int i = tid; i < 512; i += 128) {
      int r = i >> 4;
      int c = (i & 15) << 3;
      v8bf vv = *(const v8bf*)(Vb + ((size_t)(b * Sc + t0 + r) * KVW + g * HDc + c));
#pragma unroll
      for (int jj = 0; jj < 8; ++jj) Vtls[c + jj][r] = vv[jj];
    }
    asm volatile("s_wait_asynccnt 0x0" ::: "memory");
    __syncthreads();

    // scores tile 16x32 = 2 N-tiles, contraction over HD in 4 chunks
    v8f sc0 = {0, 0, 0, 0, 0, 0, 0, 0};
    v8f sc1 = {0, 0, 0, 0, 0, 0, 0, 0};
#pragma unroll
    for (int c = 0; c < 4; ++c) {
      v16bf kf0 = ldfrag(&Kls[lrow][c * 32 + kbase]);
      v16bf kf1 = ldfrag(&Kls[16 + lrow][c * 32 + kbase]);
      sc0 = WMMA_BF16(qf[c], kf0, sc0);
      sc1 = WMMA_BF16(qf[c], kf1, sc1);
    }

    // online softmax per row (row m lives in one 16-lane half, VGPR m%8)
#pragma unroll
    for (int v = 0; v < 8; ++v) {
      int row = qw + v + half8;
      float s0 = sc0[v] * scale + ((t0 + lrow) > row ? -1e9f : 0.0f);
      float s1 = sc1[v] * scale + ((t0 + 16 + lrow) > row ? -1e9f : 0.0f);
      float rmax = fmaxf(s0, s1);
#pragma unroll
      for (int off = 1; off < 16; off <<= 1)
        rmax = fmaxf(rmax, __shfl_xor(rmax, off, 32));
      float mnew = fmaxf(mrow[v], rmax);
      float p0 = __expf(s0 - mnew);
      float p1 = __expf(s1 - mnew);
      float rs = p0 + p1;
#pragma unroll
      for (int off = 1; off < 16; off <<= 1) rs += __shfl_xor(rs, off, 32);
      float corr = __expf(mrow[v] - mnew);
      lsum[v] = lsum[v] * corr + rs;
      mrow[v] = mnew;
#pragma unroll
      for (int jt = 0; jt < 8; ++jt) oacc[jt][v] *= corr;
      Pls[wave][v + half8][lrow]      = (bf16)p0;
      Pls[wave][v + half8][16 + lrow] = (bf16)p1;
    }
    __builtin_amdgcn_fence(__ATOMIC_ACQ_REL, "wavefront");  // order P store->load

    // P(16x32) @ V(32x128): one WMMA (k=32) per 16-wide d tile
    v16bf pf = ldfrag(&Pls[wave][lrow][kbase]);
#pragma unroll
    for (int jt = 0; jt < 8; ++jt) {
      v16bf vf = ldfrag(&Vtls[jt * 16 + lrow][kbase]);
      oacc[jt] = WMMA_BF16(pf, vf, oacc[jt]);
    }
    __syncthreads();
  }

  // Normalize and store attention output (bf16) for the Wo projection
#pragma unroll
  for (int jt = 0; jt < 8; ++jt) {
#pragma unroll
    for (int v = 0; v < 8; ++v) {
      int row = qw + v + half8;
      float o = oacc[jt][v] / lsum[v];
      AO[(size_t)(b * Sc + row) * Hc + h * HDc + jt * 16 + lrow] = (bf16)o;
    }
  }
}

// ---------------------------------------------------------------------------
extern "C" void kernel_launch(void* const* d_in, const int* in_sizes, int n_in,
                              void* d_out, int out_size, void* d_ws, size_t ws_size,
                              hipStream_t stream) {
  const float* x  = (const float*)d_in[0];
  // d_in[1] = causal_mask (implicit in kernel)
  const float* Wq = (const float*)d_in[2];
  const float* bq = (const float*)d_in[3];
  const float* Wk = (const float*)d_in[4];
  const float* bk = (const float*)d_in[5];
  const float* Wv = (const float*)d_in[6];
  const float* bv = (const float*)d_in[7];
  const float* Wo = (const float*)d_in[8];
  const float* bo = (const float*)d_in[9];
  float* out = (float*)d_out;

  const size_t M = (size_t)Bc * Sc;  // 4096
  char* ws = (char*)d_ws;
  bf16* Qw = (bf16*)ws;                                     // 4096x2048 bf16
  bf16* Kw = (bf16*)(ws + M * Hc * 2);                      // 4096x512
  bf16* Vw = (bf16*)(ws + M * Hc * 2 + M * KVW * 2);        // 4096x512
  bf16* AO = (bf16*)(ws + M * Hc * 2 + 2 * M * KVW * 2);    // 4096x2048

  dim3 gblk(256);
  gemm_wmma<float, bf16><<<dim3(Hc / 128, M / 128), gblk, 0, stream>>>(x, Wq, bq, Qw, Hc, Hc);
  gemm_wmma<float, bf16><<<dim3(KVW / 128, M / 128), gblk, 0, stream>>>(x, Wk, bk, Kw, KVW, Hc);
  gemm_wmma<float, bf16><<<dim3(KVW / 128, M / 128), gblk, 0, stream>>>(x, Wv, bv, Vw, KVW, Hc);
  attn_wmma<<<dim3(Sc / 64, Bc * NHc), dim3(128), 0, stream>>>(Qw, Kw, Vw, AO);
  gemm_wmma<bf16, float><<<dim3(Hc / 128, M / 128), gblk, 0, stream>>>(AO, Wo, bo, out, Hc, Hc);
}
